// VectorQuantize_87265145520598
// MI455X (gfx1250) — matrix operand order repaired and checked by reading, not compiled, not run on hardware
//
#include <hip/hip_runtime.h>

#define B_DIM   16
#define DIM     128
#define T_DIM   4096
#define NCODES  1024
#define NROWS   (B_DIM * T_DIM)        // 65536
#define QSIZE   (B_DIM * DIM * T_DIM)  // 8388608
#define EPAD    136                    // bf16 elems per LDS row (128 + 8 pad -> conflict-free)
#define NTILES  (NCODES / 16)          // 64

typedef __bf16 bf16_t;
typedef __attribute__((ext_vector_type(16))) __bf16 v16bf;
typedef __attribute__((ext_vector_type(8)))  __bf16 v8bf;
typedef __attribute__((ext_vector_type(8)))  float  v8f;
typedef __attribute__((ext_vector_type(4)))  float  v4f;

// ---------------- prep: code norms ||e||^2 + zero used flags ----------------
__global__ __launch_bounds__(256) void vq_prep(const float* __restrict__ embed,
                                               float* __restrict__ norms,
                                               float* __restrict__ used) {
    int i = blockIdx.x * 256 + threadIdx.x;
    if (i < NCODES) {
        const float* e = embed + (size_t)i * DIM;
        float s = 0.0f;
        #pragma unroll 16
        for (int d = 0; d < DIM; ++d) s = fmaf(e[d], e[d], s);
        norms[i] = s;
        used[i]  = 0.0f;
    }
}

// ---------------- main: distances (split-bf16 WMMA) + argmin + quantize ----------------
__global__ __launch_bounds__(256) void vq_main(
        const float* __restrict__ x, const float* __restrict__ embed,
        const float* __restrict__ norms,
        float* __restrict__ q_out, float* __restrict__ idx_out,
        float* __restrict__ used, float* __restrict__ partial)
{
    __shared__ __align__(16) bf16_t lds_eh[2][16 * EPAD];   // double-buffered hi tile
    __shared__ __align__(16) bf16_t lds_el[2][16 * EPAD];   // double-buffered lo tile
    __shared__ float lds_nrm[NCODES];
    __shared__ float lds_red[256];

    const int tid  = threadIdx.x;
    const int wave = tid >> 5;
    const int lane = tid & 31;
    const int grp  = lane >> 4;   // half-wave group
    const int ln16 = lane & 15;

    const int r0 = blockIdx.x * 128 + wave * 16;  // first flat row n of this wave's tile
    const int bb = r0 >> 12;                      // batch  (n / T)
    const int t0 = r0 & (T_DIM - 1);              // time   (n % T); tiles never cross b

    // ---- build A fragments (16 rows x K=128) as bf16 hi/lo, ISA 16-bit A layout ----
    // element e of v16bf at lane l: K = s*32 + 16*(e>=8) + 8*grp + (e&7), M = ln16
    v16bf a_hi[4], a_lo[4];
    #pragma unroll
    for (int s = 0; s < 4; ++s) {
        #pragma unroll
        for (int h = 0; h < 2; ++h) {
            const int k0 = s * 32 + h * 16 + grp * 8;
            #pragma unroll
            for (int j = 0; j < 8; ++j) {
                const int d = k0 + j;
                float xv = x[((size_t)(bb * DIM + d) << 12) + t0 + ln16];
                bf16_t hi = (bf16_t)xv;
                bf16_t lo = (bf16_t)(xv - (float)hi);
                a_hi[s][h * 8 + j] = hi;
                a_lo[s][h * 8 + j] = lo;
            }
        }
    }

    // stage all code norms into LDS once
    #pragma unroll
    for (int i = 0; i < NCODES / 256; ++i)
        lds_nrm[tid + i * 256] = norms[tid + i * 256];

    // staging assignment: thread -> tile row sn = tid>>4, 8 contiguous k at sk8 = (tid&15)*8
    const int sn  = tid >> 4;
    const int sk8 = (tid & 15) * 8;
    v4f s0, s1;

    // prologue: fetch tile 0 (2 x b128) and store into buffer 0 (2 x b128)
    s0 = *(const v4f*)&embed[(size_t)sn * DIM + sk8];
    s1 = *(const v4f*)&embed[(size_t)sn * DIM + sk8 + 4];
    {
        v8bf hv, lv;
        #pragma unroll
        for (int j = 0; j < 8; ++j) {
            float f = (j < 4) ? s0[j] : s1[j - 4];
            bf16_t h = (bf16_t)f;
            hv[j] = h;
            lv[j] = (bf16_t)(f - (float)h);
        }
        *(v8bf*)&lds_eh[0][sn * EPAD + sk8] = hv;
        *(v8bf*)&lds_el[0][sn * EPAD + sk8] = lv;
    }

    float best[8];
    int   bidx[8];
    #pragma unroll
    for (int v = 0; v < 8; ++v) { best[v] = 3.402823e38f; bidx[v] = 0; }

    for (int ct = 0; ct < NTILES; ++ct) {
        const int cb  = ct * 16;
        const int buf = ct & 1;

        // issue next tile's global loads early; latency hidden under WMMA below
        if (ct + 1 < NTILES) {
            const float* ep = embed + (size_t)(cb + 16 + sn) * DIM + sk8;
            s0 = *(const v4f*)ep;
            s1 = *(const v4f*)(ep + 4);
        }

        __syncthreads();  // buf stores (iter ct-1) visible; buf^1 reads (iter ct-1) done

        // ---- 12 WMMAs on 3 independent accumulator chains ----
        v8f acc_a = {}, acc_b = {}, acc_c = {};
        #pragma unroll
        for (int s = 0; s < 4; ++s) {
            // B layout (32x16 bf16): element e at lane l: K = s*32 + 16*(e>=8) + 8*grp + (e&7), N = ln16
            const int boff = ln16 * EPAD + s * 32 + grp * 8;
            v8bf bh0 = *(const v8bf*)&lds_eh[buf][boff];
            v8bf bh1 = *(const v8bf*)&lds_eh[buf][boff + 16];
            v8bf bl0 = *(const v8bf*)&lds_el[buf][boff];
            v8bf bl1 = *(const v8bf*)&lds_el[buf][boff + 16];
            v16bf bh = __builtin_shufflevector(bh0, bh1, 0,1,2,3,4,5,6,7,8,9,10,11,12,13,14,15);
            v16bf bl = __builtin_shufflevector(bl0, bl1, 0,1,2,3,4,5,6,7,8,9,10,11,12,13,14,15);
            acc_a = __builtin_amdgcn_wmma_f32_16x16x32_bf16(false, a_hi[s], false, bh, (short)0, acc_a, false, false);
            acc_b = __builtin_amdgcn_wmma_f32_16x16x32_bf16(false, a_hi[s], false, bl, (short)0, acc_b, false, false);
            acc_c = __builtin_amdgcn_wmma_f32_16x16x32_bf16(false, a_lo[s], false, bh, (short)0, acc_c, false, false);
        }

        // score = ||e||^2 - 2*(x.e) ; per lane: column code = cb + ln16, row M = v + 8*grp
        const float nrm  = lds_nrm[cb + ln16];
        const int   code = cb + ln16;
        #pragma unroll
        for (int v = 0; v < 8; ++v) {
            float dot = acc_a[v] + acc_b[v] + acc_c[v];
            float score = fmaf(-2.0f, dot, nrm);
            if (score < best[v]) { best[v] = score; bidx[v] = code; }  // ascending ct -> lowest idx on ties
        }

        // convert + store next tile into the other buffer (WAR-safe: barrier at loop top)
        if (ct + 1 < NTILES) {
            v8bf hv, lv;
            #pragma unroll
            for (int j = 0; j < 8; ++j) {
                float f = (j < 4) ? s0[j] : s1[j - 4];
                bf16_t h = (bf16_t)f;
                hv[j] = h;
                lv[j] = (bf16_t)(f - (float)h);
            }
            *(v8bf*)&lds_eh[buf ^ 1][sn * EPAD + sk8] = hv;
            *(v8bf*)&lds_el[buf ^ 1][sn * EPAD + sk8] = lv;
        }
    }

    // ---- min/argmin across the 16 lanes holding each row (first-index tie-break) ----
    #pragma unroll
    for (int v = 0; v < 8; ++v) {
        float s = best[v]; int i = bidx[v];
        #pragma unroll
        for (int m = 1; m < 16; m <<= 1) {
            float s2 = __shfl_xor(s, m, 32);
            int   i2 = __shfl_xor(i, m, 32);
            if (s2 < s || (s2 == s && i2 < i)) { s = s2; i = i2; }
        }
        best[v] = s; bidx[v] = i;
    }

    // lanes with ln16==0 (lane 0 -> rows v, lane 16 -> rows 8+v) publish indices + used flags
    if (ln16 == 0) {
        #pragma unroll
        for (int v = 0; v < 8; ++v) {
            const int n = r0 + 8 * grp + v;
            idx_out[n] = (float)bidx[v];
            used[bidx[v]] = 1.0f;   // idempotent store, race-free by value
        }
    }

    // broadcast the 16 winning codes to all lanes
    int codes[16];
    #pragma unroll
    for (int m = 0; m < 16; ++m)
        codes[m] = __shfl(bidx[m & 7], (m < 8) ? 0 : 16, 32);

    // ---- quantized gather + loss: lane owns dim d, handles full 16-element t-run ----
    float lsum = 0.0f;
    #pragma unroll
    for (int dd = 0; dd < 4; ++dd) {
        const int d = lane + dd * 32;
        const size_t base = ((size_t)(bb * DIM + d) << 12) + t0;   // x/q[b][d][t0]
        v4f xv[4], qv[4];
        #pragma unroll
        for (int q4 = 0; q4 < 4; ++q4)
            xv[q4] = *(const v4f*)&x[base + q4 * 4];
        #pragma unroll
        for (int m = 0; m < 16; ++m) {
            float ev = embed[(size_t)codes[m] * DIM + d];
            qv[m >> 2][m & 3] = ev;
            float diff = xv[m >> 2][m & 3] - ev;
            lsum = fmaf(diff, diff, lsum);
        }
        #pragma unroll
        for (int q4 = 0; q4 < 4; ++q4)
            *(v4f*)&q_out[base + q4 * 4] = qv[q4];
    }

    // deterministic block tree-reduction of the loss
    lds_red[tid] = lsum;
    __syncthreads();
    #pragma unroll
    for (int s = 128; s > 0; s >>= 1) {
        if (tid < s) lds_red[tid] += lds_red[tid + s];
        __syncthreads();
    }
    if (tid == 0) partial[blockIdx.x] = lds_red[0];
}

// ---------------- finalize: loss + utilization (deterministic) ----------------
__global__ __launch_bounds__(256) void vq_final(const float* __restrict__ ws,
                                                float* __restrict__ loss_out,
                                                float* __restrict__ util_out) {
    __shared__ float red[256];
    const float* used    = ws + 1024;
    const float* partial = ws + 2048;
    const int tid = threadIdx.x;

    red[tid] = used[tid] + used[tid + 256] + used[tid + 512] + used[tid + 768];
    __syncthreads();
    for (int s = 128; s > 0; s >>= 1) { if (tid < s) red[tid] += red[tid + s]; __syncthreads(); }
    float usum = red[0];
    __syncthreads();

    red[tid] = partial[tid] + partial[tid + 256];
    __syncthreads();
    for (int s = 128; s > 0; s >>= 1) { if (tid < s) red[tid] += red[tid + s]; __syncthreads(); }

    if (tid == 0) {
        loss_out[0] = 2.0f * red[0] / (float)((size_t)NROWS * DIM);
        util_out[0] = usum / (float)NCODES;
    }
}

extern "C" void kernel_launch(void* const* d_in, const int* in_sizes, int n_in,
                              void* d_out, int out_size, void* d_ws, size_t ws_size,
                              hipStream_t stream) {
    (void)in_sizes; (void)n_in; (void)out_size; (void)ws_size;
    const float* x     = (const float*)d_in[0];
    const float* embed = (const float*)d_in[1];

    float* out      = (float*)d_out;
    float* q_out    = out;                  // [B, DIM, T]
    float* idx_out  = out + QSIZE;          // [N] as float
    float* loss_out = out + QSIZE + NROWS;  // scalar
    float* util_out = loss_out + 1;         // scalar

    float* ws      = (float*)d_ws;
    float* norms   = ws;                    // 1024 f32
    float* used    = ws + 1024;             // 1024 f32
    float* partial = ws + 2048;             // 512  f32

    vq_prep <<<NCODES / 256, 256, 0, stream>>>(embed, norms, used);
    vq_main <<<NROWS / 128, 256, 0, stream>>>(x, embed, norms, q_out, idx_out, used, partial);
    vq_final<<<1, 256, 0, stream>>>(ws, loss_out, util_out);
}
